// SimpleLSTM_70446053589256
// MI455X (gfx1250) — matrix operand order repaired and checked by reading, not compiled, not run on hardware
//
#include <hip/hip_runtime.h>
#include <hip/hip_bf16.h>

typedef __attribute__((ext_vector_type(16))) __bf16 v16bf;
typedef __attribute__((ext_vector_type(8)))  float  v8f;

#define BB 256
#define SS 512
#define DD 768
#define HH 50
#define GG 200      // 4*H
#define GP 208      // padded gate stride (13 * 16)
#define NT 13       // N tiles of 16
#define KSTEPS 24   // 768 / 32

// ---------------------------------------------------------------------------
// Kernel 1: xg[b*S+t, g] = sum_d x[b,t,d] * W_ih[g,d] + (b_ih[g]+b_hh[g])
// One wave per 16x16 output tile; K-loop of 24 x v_wmma_f32_16x16x32_bf16.
// ---------------------------------------------------------------------------
__global__ __launch_bounds__(256) void xg_gemm_kernel(
    const float* __restrict__ x, const float* __restrict__ Wih,
    const float* __restrict__ bih, const float* __restrict__ bhh,
    float* __restrict__ xg)
{
  const int lane  = threadIdx.x & 31;
  const int wave  = threadIdx.x >> 5;
  const int tile  = blockIdx.x * 8 + wave;          // < 8192*13
  const int mt    = tile / NT;
  const int nt    = tile - mt * NT;
  const int lrow  = lane & 15;
  const int khalf = lane >> 4;

  const long arow = (long)mt * 16 + lrow;           // row of x (b*S+t)
  const float* xrow = x + arow * DD;
  const int gcol  = nt * 16 + lrow;                 // gate column
  const float* wrow = Wih + (long)(gcol < GG ? gcol : GG - 1) * DD;

  v8f acc = {};
  for (int ks = 0; ks < KSTEPS; ++ks) {
    const int kb = ks * 32;
    // ---- A fragment: elems 0..7 -> K=kb+khalf*8+e ; 8..15 -> +16
    float af[16];
    *(float4*)(af + 0)  = *(const float4*)(xrow + kb + khalf * 8);
    *(float4*)(af + 4)  = *(const float4*)(xrow + kb + khalf * 8 + 4);
    *(float4*)(af + 8)  = *(const float4*)(xrow + kb + 16 + khalf * 8);
    *(float4*)(af + 12) = *(const float4*)(xrow + kb + 16 + khalf * 8 + 4);
    v16bf a;
#pragma unroll
    for (int e = 0; e < 16; ++e) a[e] = (__bf16)af[e];
    // ---- B fragment: elems 0..7 -> K=kb+khalf*8+e ; 8..15 -> +16
    float bf[16];
    const float* bp = wrow + kb + khalf * 8;
    *(float4*)(bf + 0)  = *(const float4*)(bp + 0);
    *(float4*)(bf + 4)  = *(const float4*)(bp + 4);
    *(float4*)(bf + 8)  = *(const float4*)(bp + 16);
    *(float4*)(bf + 12) = *(const float4*)(bp + 20);
    v16bf b;
#pragma unroll
    for (int e = 0; e < 16; ++e) b[e] = (__bf16)bf[e];

    acc = __builtin_amdgcn_wmma_f32_16x16x32_bf16(
        false, a, false, b, (short)0, acc, false, false);
  }

  if (gcol < GG) {
    const float bias = bih[gcol] + bhh[gcol];
#pragma unroll
    for (int i = 0; i < 8; ++i) {
      const long row = (long)mt * 16 + khalf * 8 + i;   // D: M = i + khalf*8
      xg[row * GP + gcol] = acc[i] + bias;
    }
  }
}

// ---------------------------------------------------------------------------
// Kernel 2: per-workgroup 16-row batch slice, full 512-step recurrence.
// 8 waves; wave w owns N-tiles {w, w+8} (13 total). h kept in LDS as the
// WMMA A-fragment source (16x64 bf16, K-padded with zeros). 2 WMMAs/tile/step.
// xg and mask are register double-buffered (prefetch t+1 during step t).
// ---------------------------------------------------------------------------
__global__ __launch_bounds__(256) void lstm_rec_kernel(
    const float* __restrict__ xg, const int* __restrict__ mask,
    const float* __restrict__ Whh, const float* __restrict__ Wcls,
    const float* __restrict__ bcls, float* __restrict__ out)
{
  __shared__ __align__(16) __bf16 hA[16 * 64];
  __shared__ float gatesLds[16 * GP];

  const int tid   = threadIdx.x;
  const int lane  = tid & 31;
  const int wave  = tid >> 5;
  const int lrow  = lane & 15;
  const int khalf = lane >> 4;
  const int wgBase = blockIdx.x * 16;               // global batch row base

  // ---- W_hh^T B-fragments (K padded 50 -> 64), built once.
  v16bf bfrag[2][2] = {};
#pragma unroll
  for (int s = 0; s < 2; ++s) {
    const int nt = wave + 8 * s;
    if (nt < NT) {
      const int n = nt * 16 + lrow;                 // gate index (col)
#pragma unroll
      for (int kb2 = 0; kb2 < 2; ++kb2) {
#pragma unroll
        for (int e = 0; e < 16; ++e) {
          const int k = kb2 * 32 + (e < 8 ? khalf * 8 + e : 16 + khalf * 8 + (e - 8));  // hidden index (row of B)
          const float v = (n < GG && k < HH) ? Whh[n * HH + k] : 0.f;
          bfrag[s][kb2][e] = (__bf16)v;
        }
      }
    }
  }

  // ---- init h LDS (covers K padding too) and per-item c/h state.
  for (int i = tid; i < 16 * 64; i += 256) hA[i] = (__bf16)0.f;
  float cst[4] = {}, hst[4] = {};

  float xgbuf[2][2][8] = {};
  int   mbuf[2][4] = {};

  auto loadXg = [&](int t, int buf) {
#pragma unroll
    for (int s = 0; s < 2; ++s) {
      const int nt = wave + 8 * s;
      const int g  = nt * 16 + lrow;
      if (nt < NT && g < GG) {
#pragma unroll
        for (int i = 0; i < 8; ++i) {
          const long b = wgBase + khalf * 8 + i;
          xgbuf[buf][s][i] = xg[(b * SS + t) * GP + g];
        }
      }
    }
  };
  auto loadMask = [&](int t, int buf) {
#pragma unroll
    for (int it = 0; it < 4; ++it) {
      const int item = tid + it * 256;
      if (item < 16 * HH) {
        const int r = item / HH;
        mbuf[buf][it] = mask[(long)(wgBase + r) * SS + t];
      }
    }
  };

  loadXg(0, 0);
  loadMask(0, 0);
  __syncthreads();

  for (int t = 0; t < SS; ++t) {
    const int cur = t & 1, nxt = cur ^ 1;
    const int tn = (t + 1 < SS) ? t + 1 : t;
    loadXg(tn, nxt);                 // issue early; consumed next iteration
    loadMask(tn, nxt);

    // ---- A fragments from h (LDS), wave-local
    union { float4 q[2]; v16bf v; } afr[2];
#pragma unroll
    for (int kb2 = 0; kb2 < 2; ++kb2) {
      const __bf16* base = hA + lrow * 64 + kb2 * 32 + khalf * 8;
      afr[kb2].q[0] = *(const float4*)(base);
      afr[kb2].q[1] = *(const float4*)(base + 16);
    }

    // ---- gate GEMM + activation (wave-uniform tile loop: EXEC stays full)
#pragma unroll
    for (int s = 0; s < 2; ++s) {
      const int nt = wave + 8 * s;
      if (nt < NT) {
        v8f acc = {};
        acc = __builtin_amdgcn_wmma_f32_16x16x32_bf16(
            false, afr[0].v, false, bfrag[s][0], (short)0, acc, false, false);
        acc = __builtin_amdgcn_wmma_f32_16x16x32_bf16(
            false, afr[1].v, false, bfrag[s][1], (short)0, acc, false, false);
        const int g = nt * 16 + lrow;
        const bool isTanh = (g >= 2 * HH) && (g < 3 * HH);
#pragma unroll
        for (int i = 0; i < 8; ++i) {
          const float pre = acc[i] + xgbuf[cur][s][i];
          const float e = __expf(isTanh ? -2.f * pre : -pre);
          const float r = __builtin_amdgcn_rcpf(1.f + e);
          const float act = isTanh ? (2.f * r - 1.f) : r;
          gatesLds[(khalf * 8 + i) * GP + g] = act;
        }
      }
    }
    __syncthreads();

    // ---- elementwise LSTM cell update; write h back as bf16 A-source
#pragma unroll
    for (int it = 0; it < 4; ++it) {
      const int item = tid + it * 256;
      if (item < 16 * HH) {
        const int r = item / HH;
        const int j = item - r * HH;
        const float iv = gatesLds[r * GP + j];
        const float fv = gatesLds[r * GP + j + HH];
        const float gv = gatesLds[r * GP + j + 2 * HH];
        const float ov = gatesLds[r * GP + j + 3 * HH];
        const float cn = fv * cst[it] + iv * gv;
        const float e  = __expf(-2.f * cn);
        const float th = 2.f * __builtin_amdgcn_rcpf(1.f + e) - 1.f;
        const float hn = ov * th;
        if (mbuf[cur][it]) { cst[it] = cn; hst[it] = hn; }
        hA[r * 64 + j] = (__bf16)hst[it];
      }
    }
    __syncthreads();
  }

  // ---- classifier: h @ W_cls^T + b_cls  (reuse gatesLds as f32 h buffer)
#pragma unroll
  for (int it = 0; it < 4; ++it) {
    const int item = tid + it * 256;
    if (item < 16 * HH) {
      const int r = item / HH;
      const int j = item - r * HH;
      gatesLds[r * GP + j] = hst[it];
    }
  }
  __syncthreads();
  if (tid < 32) {
    const int r = tid >> 1, c = tid & 1;
    float sum = bcls[c];
    for (int j = 0; j < HH; ++j) sum += gatesLds[r * GP + j] * Wcls[c * HH + j];
    out[(wgBase + r) * 2 + c] = sum;
  }
}

// ---------------------------------------------------------------------------
extern "C" void kernel_launch(void* const* d_in, const int* in_sizes, int n_in,
                              void* d_out, int out_size, void* d_ws, size_t ws_size,
                              hipStream_t stream) {
  const float* x    = (const float*)d_in[0];
  const int*   mask = (const int*)d_in[1];
  const float* Wih  = (const float*)d_in[2];
  const float* Whh  = (const float*)d_in[3];
  const float* bih  = (const float*)d_in[4];
  const float* bhh  = (const float*)d_in[5];
  const float* Wcls = (const float*)d_in[6];
  const float* bcls = (const float*)d_in[7];
  float* out = (float*)d_out;
  float* xg  = (float*)d_ws;          // (B*S) x 208 fp32 = ~109 MB

  // 8192 M-tiles * 13 N-tiles = 106496 tiles, 8 waves/block -> 13312 blocks
  xg_gemm_kernel<<<13312, 256, 0, stream>>>(x, Wih, bih, bhh, xg);
  // 16 workgroups, each owns 16 batch rows for the full 512-step scan
  lstm_rec_kernel<<<BB / 16, 256, 0, stream>>>(xg, mask, Whh, Wcls, bcls, out);
}
